// MambaG2G_30425548325258
// MI455X (gfx1250) — compile-verified, matrix operand order checked by reading, not run on hardware
//
#include <hip/hip_runtime.h>
#include <hip/hip_bf16.h>
#include <math.h>
#include <stdint.h>

// ---------------------------------------------------------------------------
// MI455X / gfx1250, wave32. All matmuls via v_wmma_f32_16x16x32_f16 with
// Tensor-Data-Mover (tensor_load_to_lds) double-buffered tile staging.
// Model: per-step GINE + BatchNorm + Mamba(d=1000,expand=2) over 16 adjacency
// slices, then a top Mamba(expand=1) over the T=16 axis, then 3 output GEMMs.
// ---------------------------------------------------------------------------

typedef __attribute__((ext_vector_type(16))) _Float16 v16h;
typedef __attribute__((ext_vector_type(8)))  _Float16 v8h;
typedef __attribute__((ext_vector_type(8)))  float    v8f;
typedef __attribute__((ext_vector_type(4)))  unsigned int v4u;
typedef __attribute__((ext_vector_type(8)))  int      v8i;
typedef __attribute__((ext_vector_type(4)))  int      v4i;

#define NN 1000        // nodes
#define TT 16          // time steps
#define DM 1000        // d_model
#define NST 16         // d_state
#define DTR 63         // dt_rank = ceil(1000/16)
#define XPN 95         // dt_rank + 2*d_state
#define XPNP 96        // padded row stride for xdbl

__device__ __forceinline__ float act_apply(float v, int act) {
  switch (act) {
    case 1: return v > 0.f ? v : 0.f;                                          // relu
    case 2: { float t = tanhf(v); return t > 0.f ? t : (__expf(t) - 1.f); }    // elu(tanh(v))
    case 3: { float e = (v > 0.f) ? v : (__expf(v) - 1.f);
              return e + 1.f + 1e-14f; }                                       // elu(v)+1+eps
    case 4: return v > 20.f ? v : log1pf(__expf(v));                           // softplus
    default: return v;
  }
}
__device__ __forceinline__ float silu_f(float v) { return v / (1.f + __expf(-v)); }

// ---------------------------------------------------------------------------
// TDM 2D tile load: D# group0/group1 per CDNA5 ISA 8.3/8.4.
//   group0: [1:0]=count=1, [63:32]=lds byte addr, [120:64]=global addr,
//           [127:126]=type=2.
//   group1: [17:16]=data_size(2 -> 4B), [79:48]=tensor_dim0 (remaining cols,
//           OOB reads -> 0), [111:80]=tensor_dim1 (remaining rows),
//           [127:112]=tile_dim0, [143:128]=tile_dim1,
//           [207:160]=tensor_dim0_stride (elements).
// OOB zero-fill gives free K-tail / M,N-edge padding.
// ---------------------------------------------------------------------------
__device__ __forceinline__ void tdm_load_2d(unsigned ldsOff, const float* gptr,
                                            int dim0, int dim1, long long stride,
                                            int tile0, int tile1)
{
  const unsigned long long ga = (unsigned long long)(uintptr_t)gptr;
  const unsigned d0 = (unsigned)dim0, d1 = (unsigned)dim1;
  v4u g0;
  g0[0] = 1u;                                            // count=1, user desc
  g0[1] = ldsOff;                                        // lds_addr (bytes)
  g0[2] = (unsigned)ga;                                  // global_addr[31:0]
  g0[3] = (unsigned)((ga >> 32) & 0x1FFFFFFull) | (2u << 30);  // [56:32] | type=2
  v8i g1;
  g1[0] = (int)(2u << 16);                               // data_size = 4 bytes
  g1[1] = (int)((d0 & 0xFFFFu) << 16);                   // tensor_dim0[15:0]
  g1[2] = (int)((d0 >> 16) | ((d1 & 0xFFFFu) << 16));    // dim0[31:16] | dim1[15:0]
  g1[3] = (int)((d1 >> 16) | (((unsigned)tile0) << 16)); // dim1[31:16] | tile_dim0
  g1[4] = (int)((unsigned)tile1 & 0xFFFFu);              // tile_dim1 (tile_dim2=0)
  g1[5] = (int)(unsigned)(stride & 0xFFFFFFFFll);        // dim0_stride[31:0]
  g1[6] = (int)(unsigned)((stride >> 32) & 0xFFFFll);    // dim0_stride[47:32]
  g1[7] = 0;
  const v4i z4 = {0, 0, 0, 0};
#if __clang_major__ >= 23
  const v8i z8 = {0, 0, 0, 0, 0, 0, 0, 0};
  __builtin_amdgcn_tensor_load_to_lds(g0, g1, z4, z4, z8, 0);
#else
  __builtin_amdgcn_tensor_load_to_lds(g0, g1, z4, z4, 0);
#endif
}

// ---------------------------------------------------------------------------
// Generic batched GEMM:  C = act( op(A) @ B + bias + Radd )
// op(A)(m,k) = TA ? A[k*lda+m] : A[m*lda+k].
// Block: 128 threads (4 waves), 64x64 C tile, K chunked by 32.
// TDM DMAs f32 tiles into LDS (double buffered, s_wait_tensorcnt), a convert
// pass packs them to f16 in the WMMA fragment layout, then 4 WMMAs per wave.
// ---------------------------------------------------------------------------
template <bool TA>
__launch_bounds__(128, 2)
__global__ void k_gemm_wmma(const float* __restrict__ A, long long sAb, int lda,
                            const float* __restrict__ B, long long sBb, int ldb,
                            const float* __restrict__ Radd, long long sRb, int ldr,
                            const float* __restrict__ bias,
                            float* __restrict__ C, long long sCb, int ldc,
                            int M, int N, int K, int act)
{
  __shared__ float    fA[2][64 * 32];     // TDM staging, A tile (f32)
  __shared__ float    fB[2][64 * 32];     // TDM staging, B tile (f32), [k][n]
  __shared__ _Float16 lA[64 * 40];        // [row][k] f16, 16B-aligned rows
  __shared__ _Float16 lB[64 * 40];        // [ncol][k] f16

  const int bz = blockIdx.z;
  A += sAb * bz; B += sBb * bz; C += sCb * bz;
  if (Radd) Radd += sRb * bz;

  const int tid  = threadIdx.x;
  const int lane = tid & 31;
  const int wv   = tid >> 5;
  const int m0   = blockIdx.y * 64;
  const int n0   = blockIdx.x * 64;

  const unsigned offA0 = (unsigned)(uintptr_t)&fA[0][0];
  const unsigned offA1 = (unsigned)(uintptr_t)&fA[1][0];
  const unsigned offB0 = (unsigned)(uintptr_t)&fB[0][0];
  const unsigned offB1 = (unsigned)(uintptr_t)&fB[1][0];

  // convert-pass map: r = tid>>1 in [0,64) selects row (A) / col (B);
  // c0 = (tid&1)*16 selects k-halfchunk.
  const int r  = tid >> 1;
  const int c0 = (tid & 1) << 4;

  const int nCh = (K + 31) >> 5;

  auto issue = [&](int k0, int par) {
    if (!TA)  // A tile rows m0..m0+63, cols k0..k0+31, layout [m][k]
      tdm_load_2d(par ? offA1 : offA0, A + (long long)m0 * lda + k0,
                  K - k0, M - m0, (long long)lda, 32, 64);
    else      // A^T tile: rows k, cols m, layout [k][m]
      tdm_load_2d(par ? offA1 : offA0, A + (long long)k0 * lda + m0,
                  M - m0, K - k0, (long long)lda, 64, 32);
    // B tile rows k0..k0+31, cols n0..n0+63, layout [k][n]
    tdm_load_2d(par ? offB1 : offB0, B + (long long)k0 * ldb + n0,
                N - n0, K - k0, (long long)ldb, 64, 32);
  };

  if (wv == 0) issue(0, 0);

  v8f acc[4] = {};

  for (int ci = 0; ci < nCh; ++ci) {
    const int par = ci & 1;
    if (wv == 0) {
      if (ci + 1 < nCh) {
        issue((ci + 1) << 5, par ^ 1);
        __builtin_amdgcn_s_wait_tensorcnt(2);   // chunk ci's two DMAs done
      } else {
        __builtin_amdgcn_s_wait_tensorcnt(0);
      }
    }
    __syncthreads();   // publish DMA data; also guards f16 buffer reuse

    // ---- convert f32 staging -> f16 fragment layout ----
    float av[16], bv[16];
    if (!TA) {         // fA[par] is [m][k] (row len 32): contiguous reads
      const float4* pa = (const float4*)&fA[par][r * 32 + c0];
      #pragma unroll
      for (int q = 0; q < 4; ++q) {
        const float4 f = pa[q];
        av[q * 4 + 0] = f.x; av[q * 4 + 1] = f.y;
        av[q * 4 + 2] = f.z; av[q * 4 + 3] = f.w;
      }
    } else {           // fA[par] is [k][m] (row len 64): strided reads
      #pragma unroll
      for (int c = 0; c < 16; ++c)
        av[c] = fA[par][(c0 + c) * 64 + r];
    }
    #pragma unroll
    for (int c = 0; c < 16; ++c)        // fB[par] is [k][n]: transpose read
      bv[c] = fB[par][(c0 + c) * 64 + r];

    v8h pa0, pa1, pb0, pb1;
    #pragma unroll
    for (int q = 0; q < 8; ++q) {
      pa0[q] = (_Float16)av[q];     pa1[q] = (_Float16)av[q + 8];
      pb0[q] = (_Float16)bv[q];     pb1[q] = (_Float16)bv[q + 8];
    }
    *(v8h*)&lA[r * 40 + c0]     = pa0;
    *(v8h*)&lA[r * 40 + c0 + 8] = pa1;
    *(v8h*)&lB[r * 40 + c0]     = pb0;
    *(v8h*)&lB[r * 40 + c0 + 8] = pb1;
    __syncthreads();

    // ---- fragments + WMMA ----
    // A fragment: lane L holds row (L&15); kb = (L>=16)?8:0;
    // halves 0..7 = k in [kb,kb+8), halves 8..15 = k in [16+kb,16+kb+8)
    const int mRow = (wv << 4) + (lane & 15);
    const int kb   = (lane >> 4) << 3;
    union { v16h v; v8h h[2]; } af;
    af.h[0] = *(const v8h*)&lA[mRow * 40 + kb];
    af.h[1] = *(const v8h*)&lA[mRow * 40 + 16 + kb];

    #pragma unroll
    for (int s = 0; s < 4; ++s) {
      const int nCol = (s << 4) + (lane & 15);
      union { v16h v; v8h h[2]; } bf;
      bf.h[0] = *(const v8h*)&lB[nCol * 40 + kb];
      bf.h[1] = *(const v8h*)&lB[nCol * 40 + 16 + kb];
      acc[s] = __builtin_amdgcn_wmma_f32_16x16x32_f16(false, af.v, false, bf.v,
                                                      (short)0, acc[s], false, false);
    }
  }

  // C/D layout: lanes 0-15 -> M=r, lanes 16-31 -> M=8+r; N = n0 + 16s + (lane&15)
  const int rowBase = m0 + (wv << 4) + ((lane & 16) ? 8 : 0);
  const int nlo = lane & 15;
  #pragma unroll
  for (int s = 0; s < 4; ++s) {
    const int col = n0 + (s << 4) + nlo;
    if (col < N) {
      const float bvv = bias ? bias[col] : 0.f;
      #pragma unroll
      for (int rr = 0; rr < 8; ++rr) {
        const int row = rowBase + rr;
        if (row < M) {
          float v = acc[s][rr] + bvv;
          if (Radd) v += Radd[(long long)row * ldr + col];
          C[(long long)row * ldc + col] = act_apply(v, act);
        }
      }
    }
  }
}

// ---------------------------------------------------------------------------
// msg[t][k][n] = relu(adj[k][t][n] + emb1[n])
// ---------------------------------------------------------------------------
__global__ void k_msg(const float* __restrict__ adj, const float* __restrict__ emb1,
                      float* __restrict__ msg, int total)
{
  int idx = blockIdx.x * blockDim.x + threadIdx.x;
  if (idx >= total) return;
  int t = idx / (NN * NN);
  int r = idx - t * (NN * NN);
  int k = r / NN;
  int n = r - k * NN;
  float v = adj[(long long)k * (TT * NN) + t * NN + n] + emb1[n];
  msg[idx] = v > 0.f ? v : 0.f;
}

// ---------------------------------------------------------------------------
// BatchNorm over nodes: per (t, feature) mean / rstd, then apply in place.
// ---------------------------------------------------------------------------
__global__ void k_bn_stats(const float* __restrict__ H, float* __restrict__ mean,
                           float* __restrict__ rstd)
{
  const int col = blockIdx.x;              // t*DM + j
  const int t = col / DM, j = col - t * DM;
  __shared__ float s1[256], s2[256];
  float a = 0.f, b = 0.f;
  for (int i = threadIdx.x; i < NN; i += 256) {
    float v = H[((long long)t * NN + i) * DM + j];
    a += v; b += v * v;
  }
  s1[threadIdx.x] = a; s2[threadIdx.x] = b;
  __syncthreads();
  for (int off = 128; off > 0; off >>= 1) {
    if (threadIdx.x < off) { s1[threadIdx.x] += s1[threadIdx.x + off];
                             s2[threadIdx.x] += s2[threadIdx.x + off]; }
    __syncthreads();
  }
  if (threadIdx.x == 0) {
    float m = s1[0] / (float)NN;
    float var = s2[0] / (float)NN - m * m;
    mean[col] = m;
    rstd[col] = rsqrtf(var + 1e-5f);
  }
}

__global__ void k_bn_apply(float* __restrict__ H, const float* __restrict__ mean,
                           const float* __restrict__ rstd,
                           const float* __restrict__ gamma, const float* __restrict__ beta,
                           int total)
{
  int idx = blockIdx.x * blockDim.x + threadIdx.x;
  if (idx >= total) return;
  int t = idx / (NN * DM);
  int j = idx % DM;
  int c = t * DM + j;
  H[idx] = (H[idx] - mean[c]) * rstd[c] * gamma[j] + beta[j];
}

// ---------------------------------------------------------------------------
// RMSNorm of x (= adj slice) into U[(t*NN+i)][j]
// ---------------------------------------------------------------------------
__global__ void k_rms(const float* __restrict__ adj, const float* __restrict__ w,
                      float* __restrict__ U)
{
  const int row = blockIdx.x;               // t*NN + i
  const int t = row / NN, i = row - t * NN;
  const float* x = &adj[(long long)i * (TT * NN) + t * NN];
  __shared__ float sred[256];
  __shared__ float rs;
  float a = 0.f;
  for (int j = threadIdx.x; j < DM; j += 256) { float v = x[j]; a += v * v; }
  sred[threadIdx.x] = a;
  __syncthreads();
  for (int off = 128; off > 0; off >>= 1) {
    if (threadIdx.x < off) sred[threadIdx.x] += sred[threadIdx.x + off];
    __syncthreads();
  }
  if (threadIdx.x == 0) rs = rsqrtf(sred[0] / (float)DM + 1e-5f);
  __syncthreads();
  for (int j = threadIdx.x; j < DM; j += 256)
    U[(long long)row * DM + j] = x[j] * rs * w[j];
}

// ---------------------------------------------------------------------------
// Depthwise causal conv (d_conv=4) + SiLU.  src row stride srcStride (xin part
// of in_proj output), dst contiguous [b*L+l][Din].
// ---------------------------------------------------------------------------
__global__ void k_conv_silu(const float* __restrict__ src, int srcStride,
                            float* __restrict__ dst,
                            const float* __restrict__ cw, const float* __restrict__ cb,
                            int Bn, int L, int Din)
{
  long long idx = (long long)blockIdx.x * blockDim.x + threadIdx.x;
  long long total = (long long)Bn * L * Din;
  if (idx >= total) return;
  int d = (int)(idx % Din);
  long long rl = idx / Din;
  int l = (int)(rl % L);
  long long b = rl / L;
  float acc = cb[d];
  #pragma unroll
  for (int j = 0; j < 4; ++j) {
    int ls = l - 3 + j;
    if (ls >= 0) acc += src[(b * L + ls) * (long long)srcStride + d] * cw[d * 4 + j];
  }
  dst[(b * L + l) * (long long)Din + d] = silu_f(acc);
}

// ---------------------------------------------------------------------------
// Selective scan, one thread per (batch, channel).  16-state in registers.
// Fuses  y = (scan + D*xc) * silu(res).
// ---------------------------------------------------------------------------
__global__ void k_scan(const float* __restrict__ delta,
                       const float* __restrict__ xdbl, int ldx, int dtRank,
                       const float* __restrict__ xc,
                       const float* __restrict__ res, int resStride,
                       const float* __restrict__ A_log, const float* __restrict__ Dp,
                       float* __restrict__ y,
                       int Bn, int L, int Din)
{
  long long gid = (long long)blockIdx.x * blockDim.x + threadIdx.x;
  if (gid >= (long long)Bn * Din) return;
  const int d = (int)(gid % Din);
  const int b = (int)(gid / Din);
  float Aa[NST], h[NST];
  #pragma unroll
  for (int n = 0; n < NST; ++n) { Aa[n] = -__expf(A_log[d * NST + n]); h[n] = 0.f; }
  const float Dd = Dp[d];
  for (int l = 0; l < L; ++l) {
    const long long row = (long long)b * L + l;
    const float dt = delta[row * Din + d];
    const float xv = xc[row * Din + d];
    const float dx = dt * xv;
    const float* bc = &xdbl[row * ldx + dtRank];  // B at [0..15], C at [16..31]
    float accv = 0.f;
    #pragma unroll
    for (int n = 0; n < NST; ++n) {
      h[n] = __expf(dt * Aa[n]) * h[n] + dx * bc[n];
      accv += h[n] * bc[NST + n];
    }
    const float rv = res[row * (long long)resStride + d];
    y[row * Din + d] = (accv + Dd * xv) * silu_f(rv);
  }
}

// ---------------------------------------------------------------------------
// Z[(i*TT+t)][d] = Hbn[(t*NN+i)][d] + adj[i][t][d] + HM[(t*NN+i)][d]
// ---------------------------------------------------------------------------
__global__ void k_z(const float* __restrict__ Hb, const float* __restrict__ adj,
                    const float* __restrict__ HM, float* __restrict__ Z, int total)
{
  int idx = blockIdx.x * blockDim.x + threadIdx.x;
  if (idx >= total) return;
  int i = idx / (TT * DM);
  int rem = idx - i * (TT * DM);
  int t = rem / DM;
  int d = rem - t * DM;
  long long src = ((long long)t * NN + i) * DM + d;
  Z[idx] = Hb[src] + adj[idx] + HM[src];
}

// E[i][d] = mean_t OT[(i*TT+t)][d]
__global__ void k_mean(const float* __restrict__ OT, float* __restrict__ E, int total)
{
  int idx = blockIdx.x * blockDim.x + threadIdx.x;
  if (idx >= total) return;
  int i = idx / DM, d = idx - i * DM;
  float s = 0.f;
  #pragma unroll
  for (int t = 0; t < TT; ++t) s += OT[((long long)i * TT + t) * DM + d];
  E[idx] = s * (1.f / (float)TT);
}

// ---------------------------------------------------------------------------
extern "C" void kernel_launch(void* const* d_in, const int* in_sizes, int n_in,
                              void* d_out, int out_size, void* d_ws, size_t ws_size,
                              hipStream_t stream)
{
  (void)in_sizes; (void)n_in; (void)out_size; (void)ws_size;

  // setup_inputs() dict order, recursing into params in insertion order
  const float* adj   = (const float*)d_in[0];
  const float* emb1  = (const float*)d_in[1] + DM;      // edge_emb row 1
  const float* gw1   = (const float*)d_in[2];
  const float* gb1   = (const float*)d_in[3];
  const float* gw2   = (const float*)d_in[4];
  const float* gb2   = (const float*)d_in[5];
  const float* gamma = (const float*)d_in[6];
  const float* beta  = (const float*)d_in[7];
  const float* mqw   = (const float*)d_in[8];
  // mamba_q (d_inner = 2000)
  const float* q_in  = (const float*)d_in[9];
  const float* q_cw  = (const float*)d_in[10];
  const float* q_cb  = (const float*)d_in[11];
  const float* q_xp  = (const float*)d_in[12];
  const float* q_dtw = (const float*)d_in[13];
  const float* q_dtb = (const float*)d_in[14];
  const float* q_Al  = (const float*)d_in[15];
  const float* q_D   = (const float*)d_in[16];
  const float* q_op  = (const float*)d_in[17];
  // mamba_top (d_inner = 1000)
  const float* t_in  = (const float*)d_in[18];
  const float* t_cw  = (const float*)d_in[19];
  const float* t_cb  = (const float*)d_in[20];
  const float* t_xp  = (const float*)d_in[21];
  const float* t_dtw = (const float*)d_in[22];
  const float* t_dtb = (const float*)d_in[23];
  const float* t_Al  = (const float*)d_in[24];
  const float* t_D   = (const float*)d_in[25];
  const float* t_op  = (const float*)d_in[26];
  const float* outw  = (const float*)d_in[27];
  const float* outb  = (const float*)d_in[28];
  const float* muw   = (const float*)d_in[29];
  const float* mub   = (const float*)d_in[30];
  const float* sgw   = (const float*)d_in[31];
  const float* sgb   = (const float*)d_in[32];

  float* out = (float*)d_out;           // [x 1e6 | mu 64000 | sigma 64000]
  float* ws  = (float*)d_ws;

  // workspace layout (floats); regions reused across phases. ~903 MB total.
  float* MSG = ws + 0LL;                // 16M  msg -> HM -> E
  float* Hb  = ws + 16000000LL;         // 16M  h/hbn -> OUT_TOP
  float* H1  = ws + 32000000LL;         // 16M  h1 -> U
  float* XR  = ws + 48000000LL;         // 64M  in_proj out (both mambas)
  float* XC  = ws + 112000000LL;        // 32M  conv out
  float* DEL = ws + 144000000LL;        // 32M  delta
  float* Yb  = ws + 176000000LL;        // 32M  scan out
  float* XDB = ws + 208000000LL;        // 1.6M xdbl (row stride 96)
  float* Zb  = ws + 209600000LL;        // 16M  z, (i,t,d) layout
  float* BNM = ws + 225600000LL;        // 16K  bn mean
  float* BNR = ws + 225616000LL;        // 16K  bn rstd

  auto gemm = [&](const float* A, long long sAb, int lda, int tA,
                  const float* B, long long sBb, int ldb,
                  const float* R, long long sRb, int ldr,
                  const float* bias, float* C, long long sCb, int ldc,
                  int M, int N, int K, int act, int batch) {
    dim3 g((unsigned)((N + 63) / 64), (unsigned)((M + 63) / 64), (unsigned)batch);
    if (tA)
      k_gemm_wmma<true><<<g, dim3(128), 0, stream>>>(A, sAb, lda, B, sBb, ldb,
                                                     R, sRb, ldr, bias, C, sCb, ldc,
                                                     M, N, K, act);
    else
      k_gemm_wmma<false><<<g, dim3(128), 0, stream>>>(A, sAb, lda, B, sBb, ldb,
                                                      R, sRb, ldr, bias, C, sCb, ldc,
                                                      M, N, K, act);
  };
  const int EW = 256;
  auto blocks = [&](long long n) { return (unsigned)((n + EW - 1) / EW); };

  // ---------------- phase 1: GINE + BN (batched over t) --------------------
  k_msg<<<blocks(16000000), EW, 0, stream>>>(adj, emb1, MSG, 16000000);
  // h = A_t^T @ msg_t + x
  gemm(adj, 1000, TT * NN, 1, MSG, 1000000, NN, adj, 1000, TT * NN,
       nullptr, Hb, 1000000, DM, NN, DM, NN, 0, TT);
  // h1 = relu(h @ w1 + b1)   (flattened M = 16000)
  gemm(Hb, 0, DM, 0, gw1, 0, DM, nullptr, 0, 0, gb1, H1, 0, DM,
       TT * NN, DM, DM, 1, 1);
  // h = h1 @ w2 + b2 + x     (batched for the strided residual)
  gemm(H1, 1000000, DM, 0, gw2, 0, DM, adj, 1000, TT * NN, gb2,
       Hb, 1000000, DM, NN, DM, DM, 0, TT);
  k_bn_stats<<<TT * DM, EW, 0, stream>>>(Hb, BNM, BNR);
  k_bn_apply<<<blocks(16000000), EW, 0, stream>>>(Hb, BNM, BNR, gamma, beta, 16000000);

  // ---------------- phase 2: mamba_q over nodes (L=1000, d_inner=2000) -----
  k_rms<<<TT * NN, EW, 0, stream>>>(adj, mqw, H1 /* U */);
  gemm(H1, 0, DM, 0, q_in, 0, 4000, nullptr, 0, 0, nullptr, XR, 0, 4000,
       TT * NN, 4000, DM, 0, 1);                                    // xr
  k_conv_silu<<<blocks(32000000LL), EW, 0, stream>>>(XR, 4000, XC, q_cw, q_cb,
                                                     TT, NN, 2000);
  gemm(XC, 0, 2000, 0, q_xp, 0, XPN, nullptr, 0, 0, nullptr, XDB, 0, XPNP,
       TT * NN, XPN, 2000, 0, 1);                                   // xdbl
  gemm(XDB, 0, XPNP, 0, q_dtw, 0, 2000, nullptr, 0, 0, q_dtb, DEL, 0, 2000,
       TT * NN, 2000, DTR, 4, 1);                                   // delta = softplus
  k_scan<<<blocks(16LL * 2000), EW, 0, stream>>>(DEL, XDB, XPNP, DTR, XC,
                                                 XR + 2000, 4000, q_Al, q_D,
                                                 Yb, TT, NN, 2000);
  gemm(Yb, 0, 2000, 0, q_op, 0, DM, nullptr, 0, 0, nullptr, MSG /* HM */, 0, DM,
       TT * NN, DM, 2000, 0, 1);                                    // out_proj
  k_z<<<blocks(16000000), EW, 0, stream>>>(Hb, adj, MSG, Zb, 16000000);

  // ---------------- phase 3: mamba_top over time (L=16, d_inner=1000) ------
  gemm(Zb, 0, DM, 0, t_in, 0, 2000, nullptr, 0, 0, nullptr, XR, 0, 2000,
       NN * TT, 2000, DM, 0, 1);                                    // xr_top
  k_conv_silu<<<blocks(16000000LL), EW, 0, stream>>>(XR, 2000, XC, t_cw, t_cb,
                                                     NN, TT, 1000);
  gemm(XC, 0, 1000, 0, t_xp, 0, XPN, nullptr, 0, 0, nullptr, XDB, 0, XPNP,
       NN * TT, XPN, 1000, 0, 1);
  gemm(XDB, 0, XPNP, 0, t_dtw, 0, 1000, nullptr, 0, 0, t_dtb, DEL, 0, 1000,
       NN * TT, 1000, DTR, 4, 1);
  k_scan<<<blocks(1000LL * 1000), EW, 0, stream>>>(DEL, XDB, XPNP, DTR, XC,
                                                   XR + 1000, 2000, t_Al, t_D,
                                                   Yb, NN, TT, 1000);
  gemm(Yb, 0, 1000, 0, t_op, 0, DM, nullptr, 0, 0, nullptr, Hb /* OUT_TOP */, 0, DM,
       NN * TT, DM, 1000, 0, 1);
  k_mean<<<blocks(1000000), EW, 0, stream>>>(Hb, MSG /* E */, 1000000);

  // ---------------- heads --------------------------------------------------
  gemm(MSG, 0, DM, 0, outw, 0, DM, nullptr, 0, 0, outb, out, 0, DM,
       NN, DM, DM, 2, 1);                                           // x = elu(tanh(.))
  gemm(out, 0, DM, 0, muw, 0, 64, nullptr, 0, 0, mub, out + 1000000, 0, 64,
       NN, 64, DM, 0, 1);                                           // mu
  gemm(out, 0, DM, 0, sgw, 0, 64, nullptr, 0, 0, sgb, out + 1064000, 0, 64,
       NN, 64, DM, 3, 1);                                           // sigma
}
